// MultiHeadSelfAttention_68736656605828
// MI455X (gfx1250) — matrix operand order repaired and checked by reading, not compiled, not run on hardware
//
#include <hip/hip_runtime.h>
#include <hip/hip_bf16.h>
#include <stdint.h>

// ---------------- problem constants ----------------
#define BATCH 4
#define SEQ   2048
#define EMB   1024
#define HEADS 16
#define HD    64
#define MTOT  (BATCH*SEQ)   // 8192

// ---------------- WMMA vector types ----------------
typedef __attribute__((ext_vector_type(16))) __bf16 v16bf;
typedef __attribute__((ext_vector_type(8)))  float  v8f;
typedef __attribute__((ext_vector_type(4)))  int    v4i;

union Frag {            // 32 bytes: one WMMA bf16 operand per lane
    v16bf v;
    uint4 u[2];
};

__device__ __forceinline__ unsigned short f2bf(float f) {
    union { float f; uint32_t u; } x; x.f = f;
    uint32_t u = x.u;
    u += 0x7FFFu + ((u >> 16) & 1u);        // round-to-nearest-even
    return (unsigned short)(u >> 16);
}

#define WMMA_BF16(A, B, C) \
    __builtin_amdgcn_wmma_f32_16x16x32_bf16(false, (A), false, (B), (short)0, (C), false, false)

// ---------------- async copy to LDS (gfx1250 path) ----------------
#if __has_builtin(__builtin_amdgcn_global_load_async_to_lds_b128)
#define HAS_ASYNC_LDS 1
#else
#define HAS_ASYNC_LDS 0
#endif

#if HAS_ASYNC_LDS
typedef __attribute__((address_space(1))) v4i* gv4i_p;   // global int4*
typedef __attribute__((address_space(3))) v4i* lv4i_p;   // LDS int4*
#endif

// copy 16 bytes global -> LDS (async when available)
__device__ __forceinline__ void cp16_to_lds(unsigned short* l, const unsigned short* g) {
#if HAS_ASYNC_LDS
    __builtin_amdgcn_global_load_async_to_lds_b128(
        (gv4i_p)(g), (lv4i_p)(l), 0, 0);
#else
    *(uint4*)l = *(const uint4*)g;
#endif
}

__device__ __forceinline__ void async_wait_all() {
#if HAS_ASYNC_LDS
  #if __has_builtin(__builtin_amdgcn_s_wait_asynccnt)
    __builtin_amdgcn_s_wait_asynccnt(0);
  #else
    asm volatile("s_wait_asynccnt 0x0" ::: "memory");
  #endif
#endif
}

// ---------------- conversion / layout kernels ----------------
__global__ void k_cvt_x(const float* __restrict__ x, unsigned short* __restrict__ xb, int n) {
    int i = blockIdx.x * blockDim.x + threadIdx.x;
    if (i < n) xb[i] = f2bf(x[i]);
}

// W[h,e,d] -> Wt[n=h*64+d][e]  (transposed so it is the WMMA B^T operand)
__global__ void k_cvt_wqkv(const float* __restrict__ w, unsigned short* __restrict__ wt) {
    int i = blockIdx.x * blockDim.x + threadIdx.x;
    if (i >= HEADS * EMB * HD) return;
    int d = i % HD;
    int e = (i / HD) % EMB;
    int h = i / (HD * EMB);
    int n = h * HD + d;
    wt[(size_t)n * EMB + e] = f2bf(w[i]);
}

// W_o[e,o] -> Wot[o][e]
__global__ void k_cvt_wo(const float* __restrict__ w, unsigned short* __restrict__ wt) {
    int i = blockIdx.x * blockDim.x + threadIdx.x;
    if (i >= EMB * EMB) return;
    int o = i % EMB;
    int e = i / EMB;
    wt[(size_t)o * EMB + e] = f2bf(w[i]);
}

// ---------------- QKV projection GEMM ----------------
// C[m,n] = sum_k A[m,k] * Bt[n,k];  m = b*2048+l, n = h*64+d
// mode 0: Q (scaled, [b,h,l,d])  1: K ([b,h,l,d])  2: V ([b,h,d,l])
__global__ __launch_bounds__(256) void k_gemm_qkv(
    const unsigned short* __restrict__ A,   // [MTOT, EMB] bf16
    const unsigned short* __restrict__ Bt,  // [EMB, EMB] bf16 (n-major)
    unsigned short* __restrict__ Out, int mode)
{
    const int K = EMB;
    const int lane = threadIdx.x & 31;
    const int w    = threadIdx.x >> 5;
    const int wm   = w & 3, wn = w >> 2;
    const int m_base = blockIdx.y * 128 + wm * 32;
    const int n_base = blockIdx.x * 64  + wn * 32;
    const int col  = lane & 15, half = lane >> 4;
    const int offA = half * 8;     // A layout: K interleave granularity 8
    const int offB = half * 8;     // B layout: K interleave granularity 8

    v8f acc[2][2];
    #pragma unroll
    for (int i = 0; i < 2; i++)
        #pragma unroll
        for (int j = 0; j < 2; j++) acc[i][j] = {};

    for (int k0 = 0; k0 < K; k0 += 32) {
        Frag af[2], bfg[2];
        #pragma unroll
        for (int i = 0; i < 2; i++) {
            const unsigned short* p = A + (size_t)(m_base + i * 16 + col) * K + k0;
            af[i].u[0] = *(const uint4*)(p + offA);
            af[i].u[1] = *(const uint4*)(p + 16 + offA);
        }
        #pragma unroll
        for (int j = 0; j < 2; j++) {
            const unsigned short* p = Bt + (size_t)(n_base + j * 16 + col) * K + k0 + offB;
            bfg[j].u[0] = *(const uint4*)(p);
            bfg[j].u[1] = *(const uint4*)(p + 16);
        }
        #pragma unroll
        for (int i = 0; i < 2; i++)
            #pragma unroll
            for (int j = 0; j < 2; j++)
                acc[i][j] = WMMA_BF16(af[i].v, bfg[j].v, acc[i][j]);
    }

    // fold softmax 1/sqrt(64) and log2(e) into Q
    const float qscale = 0.125f * 1.44269504088896340736f;
    #pragma unroll
    for (int i = 0; i < 2; i++)
        #pragma unroll
        for (int j = 0; j < 2; j++)
            #pragma unroll
            for (int r = 0; r < 8; r++) {
                int m = m_base + i * 16 + r + 8 * half;
                int n = n_base + j * 16 + col;
                float v = acc[i][j][r];
                int b = m >> 11, l = m & 2047, h = n >> 6, d = n & 63;
                size_t idx;
                if (mode == 2) idx = ((size_t)(b * HEADS + h) * HD + d) * SEQ + l;   // V^T
                else           idx = ((size_t)(b * HEADS + h) * SEQ + l) * HD + d;   // Q,K
                if (mode == 0) v *= qscale;
                Out[idx] = f2bf(v);
            }
}

// ---------------- flash attention ----------------
// grid: (SEQ/64, HEADS, BATCH), 128 threads (4 waves). Wave w owns 16 Q rows.
// K/V tiles staged into LDS once per block via async-load-to-LDS, double-buffered
// so tile i+1 streams in (ASYNCcnt) while tile i is consumed by WMMA + softmax.
__global__ __launch_bounds__(128) void k_attn(
    const unsigned short* __restrict__ Q,   // [B,H,L,D] (pre-scaled)
    const unsigned short* __restrict__ Km,  // [B,H,L,D]
    const unsigned short* __restrict__ Vt,  // [B,H,D,L]
    unsigned short* __restrict__ Oc)        // [B,L,H*D]
{
    __shared__ __align__(16) unsigned short Klds[2][64 * 64];   // [s_local][d]
    __shared__ __align__(16) unsigned short Vlds[2][64 * 64];   // [d][s_local]
    __shared__ __align__(16) unsigned short Plds[4][16][64];

    const int tid  = threadIdx.x;
    const int lane = tid & 31;
    const int w    = tid >> 5;
    const int col  = lane & 15, half = lane >> 4;
    const int offA = half * 8, offB = half * 8;
    const int l0 = blockIdx.x * 64;
    const int h  = blockIdx.y;
    const int b  = blockIdx.z;

    const unsigned short* Qbh = Q  + (size_t)(b * HEADS + h) * SEQ * HD;
    const unsigned short* Kbh = Km + (size_t)(b * HEADS + h) * SEQ * HD;
    const unsigned short* Vbh = Vt + (size_t)(b * HEADS + h) * HD * SEQ;

    // stage one 64x64 K tile + 64x64 V^T tile into LDS (4 x 16B chunks per
    // thread per tile; 128 threads cover 64 rows x 128 B)
    auto stage = [&](int buf, int s_base) {
        #pragma unroll
        for (int i = 0; i < 4; i++) {
            int chunk = tid + i * 128;      // 0..511
            int row   = chunk >> 3;         // 0..63
            int co    = (chunk & 7) * 8;    // element offset in row
            cp16_to_lds(&Klds[buf][row * 64 + co],
                        Kbh + (size_t)(s_base + row) * HD + co);
            cp16_to_lds(&Vlds[buf][row * 64 + co],
                        Vbh + (size_t)row * SEQ + s_base + co);
        }
    };

    // Q fragments (A operand), resident for whole kernel
    Frag qf[2];
    {
        const unsigned short* p = Qbh + (size_t)(l0 + w * 16 + col) * HD;
        #pragma unroll
        for (int ks = 0; ks < 2; ks++) {
            qf[ks].u[0] = *(const uint4*)(p + ks * 32 + offA);
            qf[ks].u[1] = *(const uint4*)(p + ks * 32 + 16 + offA);
        }
    }

    v8f oacc[4];
    #pragma unroll
    for (int t = 0; t < 4; t++) oacc[t] = {};
    float row_m[8], row_l[8];
    #pragma unroll
    for (int r = 0; r < 8; r++) { row_m[r] = -INFINITY; row_l[r] = 0.0f; }

    stage(0, 0);    // prologue: tile 0 in flight

    for (int s0 = 0; s0 < SEQ; s0 += 64) {
        const int buf = (s0 >> 6) & 1;
        async_wait_all();       // our async loads for tile `buf` complete
        __syncthreads();        // ...and everyone's are visible; prev buf free

        if (s0 + 64 < SEQ) stage(buf ^ 1, s0 + 64);   // overlap next tile

        const unsigned short* Kt = &Klds[buf][0];
        const unsigned short* Vb2 = &Vlds[buf][0];

        // ---- S tile = Q * K^T (16 x 64) ----
        v8f sacc[4];
        #pragma unroll
        for (int t = 0; t < 4; t++) sacc[t] = {};
        #pragma unroll
        for (int t = 0; t < 4; t++) {
            Frag kb0, kb1;
            const unsigned short* p = Kt + (size_t)(t * 16 + col) * 64;
            kb0.u[0] = *(const uint4*)(p + offB);
            kb0.u[1] = *(const uint4*)(p + offB + 16);
            kb1.u[0] = *(const uint4*)(p + 32 + offB);
            kb1.u[1] = *(const uint4*)(p + 32 + offB + 16);
            sacc[t] = WMMA_BF16(qf[0].v, kb0.v, sacc[t]);
            sacc[t] = WMMA_BF16(qf[1].v, kb1.v, sacc[t]);
        }

        // ---- online softmax (row r lives at acc index r, 16 lanes/half) ----
        #pragma unroll
        for (int r = 0; r < 8; r++) {
            float mx = fmaxf(fmaxf(sacc[0][r], sacc[1][r]), fmaxf(sacc[2][r], sacc[3][r]));
            #pragma unroll
            for (int ofs = 1; ofs < 16; ofs <<= 1) mx = fmaxf(mx, __shfl_xor(mx, ofs, 32));
            float mnew  = fmaxf(row_m[r], mx);
            float alpha = exp2f(row_m[r] - mnew);   // exp2(-inf)=0 on first tile
            float psum = 0.0f;
            #pragma unroll
            for (int t = 0; t < 4; t++) {
                float pv = exp2f(sacc[t][r] - mnew);
                sacc[t][r] = pv;
                psum += pv;
            }
            #pragma unroll
            for (int ofs = 1; ofs < 16; ofs <<= 1) psum += __shfl_xor(psum, ofs, 32);
            row_l[r] = row_l[r] * alpha + psum;
            row_m[r] = mnew;
            #pragma unroll
            for (int t = 0; t < 4; t++) oacc[t][r] *= alpha;
        }

        // ---- P: C-layout -> LDS -> A-layout (wave-local region; DS is
        //      in-order per wave, just stop the compiler reordering) ----
        #pragma unroll
        for (int t = 0; t < 4; t++)
            #pragma unroll
            for (int r = 0; r < 8; r++)
                Plds[w][r + 8 * half][t * 16 + col] = f2bf(sacc[t][r]);
        asm volatile("" ::: "memory");

        Frag pf[2];
        {
            const unsigned short* p = &Plds[w][col][0];
            #pragma unroll
            for (int ks = 0; ks < 2; ks++) {
                pf[ks].u[0] = *(const uint4*)(p + ks * 32 + offA);
                pf[ks].u[1] = *(const uint4*)(p + ks * 32 + 16 + offA);
            }
        }

        // ---- O += P * V ----
        #pragma unroll
        for (int t = 0; t < 4; t++) {
            Frag vb0, vb1;
            const unsigned short* p = Vb2 + (size_t)(t * 16 + col) * 64;
            vb0.u[0] = *(const uint4*)(p + offB);
            vb0.u[1] = *(const uint4*)(p + offB + 16);
            vb1.u[0] = *(const uint4*)(p + 32 + offB);
            vb1.u[1] = *(const uint4*)(p + 32 + offB + 16);
            oacc[t] = WMMA_BF16(pf[0].v, vb0.v, oacc[t]);
            oacc[t] = WMMA_BF16(pf[1].v, vb1.v, oacc[t]);
        }
    }

    // ---- normalize + store concat layout [b, l, h*64+d] ----
    #pragma unroll
    for (int t = 0; t < 4; t++)
        #pragma unroll
        for (int r = 0; r < 8; r++) {
            int m = l0 + w * 16 + r + 8 * half;
            float v = oacc[t][r] / row_l[r];
            Oc[((size_t)(b * SEQ + m)) * EMB + h * HD + t * 16 + col] = f2bf(v);
        }
}

// ---------------- output projection GEMM (fp32 out) ----------------
__global__ __launch_bounds__(256) void k_gemm_out(
    const unsigned short* __restrict__ A,   // Oc [MTOT, EMB]
    const unsigned short* __restrict__ Bt,  // Wot [EMB, EMB]
    float* __restrict__ Out)
{
    const int K = EMB;
    const int lane = threadIdx.x & 31;
    const int w    = threadIdx.x >> 5;
    const int wm   = w & 3, wn = w >> 2;
    const int m_base = blockIdx.y * 128 + wm * 32;
    const int n_base = blockIdx.x * 64  + wn * 32;
    const int col  = lane & 15, half = lane >> 4;
    const int offA = half * 8;
    const int offB = half * 8;

    v8f acc[2][2];
    #pragma unroll
    for (int i = 0; i < 2; i++)
        #pragma unroll
        for (int j = 0; j < 2; j++) acc[i][j] = {};

    for (int k0 = 0; k0 < K; k0 += 32) {
        Frag af[2], bfg[2];
        #pragma unroll
        for (int i = 0; i < 2; i++) {
            const unsigned short* p = A + (size_t)(m_base + i * 16 + col) * K + k0;
            af[i].u[0] = *(const uint4*)(p + offA);
            af[i].u[1] = *(const uint4*)(p + 16 + offA);
        }
        #pragma unroll
        for (int j = 0; j < 2; j++) {
            const unsigned short* p = Bt + (size_t)(n_base + j * 16 + col) * K + k0 + offB;
            bfg[j].u[0] = *(const uint4*)(p);
            bfg[j].u[1] = *(const uint4*)(p + 16);
        }
        #pragma unroll
        for (int i = 0; i < 2; i++)
            #pragma unroll
            for (int j = 0; j < 2; j++)
                acc[i][j] = WMMA_BF16(af[i].v, bfg[j].v, acc[i][j]);
    }

    #pragma unroll
    for (int i = 0; i < 2; i++)
        #pragma unroll
        for (int j = 0; j < 2; j++)
            #pragma unroll
            for (int r = 0; r < 8; r++) {
                int m = m_base + i * 16 + r + 8 * half;
                int n = n_base + j * 16 + col;
                Out[(size_t)m * EMB + n] = acc[i][j][r];
            }
}

// ---------------- launch ----------------
extern "C" void kernel_launch(void* const* d_in, const int* in_sizes, int n_in,
                              void* d_out, int out_size, void* d_ws, size_t ws_size,
                              hipStream_t stream) {
    const float* x  = (const float*)d_in[0];
    const float* Wq = (const float*)d_in[1];
    const float* Wk = (const float*)d_in[2];
    const float* Wv = (const float*)d_in[3];
    const float* Wo = (const float*)d_in[4];
    float* out = (float*)d_out;

    char* ws = (char*)d_ws;
    unsigned short* Xb  = (unsigned short*)(ws + ((size_t) 0 << 20));  // 16 MB
    unsigned short* Wqt = (unsigned short*)(ws + ((size_t)16 << 20));  //  2 MB
    unsigned short* Wkt = (unsigned short*)(ws + ((size_t)18 << 20));  //  2 MB
    unsigned short* Wvt = (unsigned short*)(ws + ((size_t)20 << 20));  //  2 MB
    unsigned short* Wot = (unsigned short*)(ws + ((size_t)22 << 20));  //  2 MB
    unsigned short* Qb  = (unsigned short*)(ws + ((size_t)24 << 20));  // 16 MB
    unsigned short* Kb  = (unsigned short*)(ws + ((size_t)40 << 20));  // 16 MB
    unsigned short* Vb  = (unsigned short*)(ws + ((size_t)56 << 20));  // 16 MB
    unsigned short* Oc  = (unsigned short*)(ws + ((size_t)72 << 20));  // 16 MB

    const int nx = MTOT * EMB;
    k_cvt_x<<<(nx + 255) / 256, 256, 0, stream>>>(x, Xb, nx);
    const int nw = HEADS * EMB * HD;
    k_cvt_wqkv<<<(nw + 255) / 256, 256, 0, stream>>>(Wq, Wqt);
    k_cvt_wqkv<<<(nw + 255) / 256, 256, 0, stream>>>(Wk, Wkt);
    k_cvt_wqkv<<<(nw + 255) / 256, 256, 0, stream>>>(Wv, Wvt);
    k_cvt_wo<<<(EMB * EMB + 255) / 256, 256, 0, stream>>>(Wo, Wot);

    dim3 g1(EMB / 64, MTOT / 128);            // (16, 64)
    k_gemm_qkv<<<g1, 256, 0, stream>>>(Xb, Wqt, Qb, 0);
    k_gemm_qkv<<<g1, 256, 0, stream>>>(Xb, Wkt, Kb, 1);
    k_gemm_qkv<<<g1, 256, 0, stream>>>(Xb, Wvt, Vb, 2);

    dim3 g2(SEQ / 64, HEADS, BATCH);          // (32, 16, 4)
    k_attn<<<g2, 128, 0, stream>>>(Qb, Kb, Vb, Oc);

    k_gemm_out<<<g1, 256, 0, stream>>>(Oc, Wot, out);
}
